// ModularPathwayConv_59794534695178
// MI455X (gfx1250) — matrix operand order repaired and checked by reading, not compiled, not run on hardware
//
#include <hip/hip_runtime.h>
#include <hip/hip_bf16.h>

typedef __attribute__((ext_vector_type(16))) __bf16 v16bf;
typedef __attribute__((ext_vector_type(8)))  __bf16 v8bf;
typedef __attribute__((ext_vector_type(8)))  float  v8f;

#define IN_CH   128
#define TWO_IN  256
#define OUT_CH  128
#define TILE_E  32              // edges per block iteration
#define SUBT    2               // two 16-row WMMA subtiles
#define NWAVES  8
#define BLOCK_T (NWAVES * 32)

#if __has_builtin(__builtin_amdgcn_sched_barrier)
#define SCHED_FENCE() __builtin_amdgcn_sched_barrier(0)
#else
#define SCHED_FENCE() asm volatile("" ::: "memory")
#endif

__global__ void zero_f32_kernel(float* __restrict__ p, int n) {
    int i = blockIdx.x * blockDim.x + threadIdx.x;
    if (i < n) p[i] = 0.0f;
}

__global__ void __launch_bounds__(BLOCK_T)
edge_mlp_wmma_kernel(const float* __restrict__ x,
                     const int*   __restrict__ edge_index,  // [2, E]
                     const float* __restrict__ edge_attr,   // [E]
                     const float* __restrict__ W1,          // [256,128] K-major
                     const float* __restrict__ b1,          // [128]
                     const float* __restrict__ W2,          // [128,128]
                     const float* __restrict__ b2,          // [128]
                     float* __restrict__ out,               // [N,128]
                     int E, int n_iters)
{
    __shared__ __align__(16) __bf16 A_ls[TILE_E][TWO_IN];   // 16 KB
    __shared__ __align__(16) __bf16 h_ls[TILE_E][OUT_CH];   //  8 KB
    __shared__ int   s_row[TILE_E];
    __shared__ int   s_col[2][TILE_E];   // double-buffered: read post-B3
    __shared__ float s_ea[TILE_E];

    const int tid  = threadIdx.x;
    const int wave = tid >> 5;
    const int lane = tid & 31;
    const int lmod = lane & 15;      // A: row M; B/C/D: N within tile
    const int lhi  = lane >> 4;      // half-wave select
    const int N    = wave * 16 + lmod;

    // ---- One-time: this wave's W1/W2 B-fragments in registers (bf16) ----
    // B layout (32x16, 16-bit): lane L -> column N=L%16; halves j=0..15 are
    // K = kstep*32 + 8*(L/16) + j + (j & 8).
    v16bf w1f[8];
    #pragma unroll
    for (int k = 0; k < 8; ++k) {
        v16bf f;
        #pragma unroll
        for (int j = 0; j < 16; ++j)
            f[j] = (__bf16)W1[(k * 32 + lhi * 8 + j + (j & 8)) * OUT_CH + N];
        w1f[k] = f;
    }
    v16bf w2f[4];
    #pragma unroll
    for (int k = 0; k < 4; ++k) {
        v16bf f;
        #pragma unroll
        for (int j = 0; j < 16; ++j)
            f[j] = (__bf16)W2[(k * 32 + lhi * 8 + j + (j & 8)) * OUT_CH + N];
        w2f[k] = f;
    }
    const float bias1 = b1[N];
    const float bias2 = b2[N];

    int parity = 0;
    for (int it = blockIdx.x; it < n_iters; it += gridDim.x, parity ^= 1) {
        const int ebase = it * TILE_E;

        if (tid < TILE_E) {
            int e  = ebase + tid;
            int ec = (e < E) ? e : (E - 1);
            s_row[tid]         = edge_index[ec];
            s_col[parity][tid] = edge_index[E + ec];
            s_ea[tid]          = (e < E) ? edge_attr[ec] : 0.0f;
        }
        __syncthreads();   // B1

        // ---- Build A tile: A[m][0:128] = x[row]*ea ; A[m][128:256] = x[col] ----
        #pragma unroll
        for (int s = 0; s < 2; ++s) {
            int id   = tid + s * BLOCK_T;       // 0..511
            int m    = id >> 4;                 // 0..31
            int cseg = (id & 15) * 16;          // 16 consecutive concat channels
            const float* src;
            float scale;
            if (cseg < IN_CH) {
                src   = x + (size_t)s_row[m] * IN_CH + cseg;
                scale = s_ea[m];
            } else {
                src   = x + (size_t)s_col[parity][m] * IN_CH + (cseg - IN_CH);
                scale = 1.0f;
            }
            float4 a0 = *(const float4*)(src + 0);
            float4 a1 = *(const float4*)(src + 4);
            float4 a2 = *(const float4*)(src + 8);
            float4 a3 = *(const float4*)(src + 12);
            v8bf lo, hi;
            lo[0]=(__bf16)(a0.x*scale); lo[1]=(__bf16)(a0.y*scale);
            lo[2]=(__bf16)(a0.z*scale); lo[3]=(__bf16)(a0.w*scale);
            lo[4]=(__bf16)(a1.x*scale); lo[5]=(__bf16)(a1.y*scale);
            lo[6]=(__bf16)(a1.z*scale); lo[7]=(__bf16)(a1.w*scale);
            hi[0]=(__bf16)(a2.x*scale); hi[1]=(__bf16)(a2.y*scale);
            hi[2]=(__bf16)(a2.z*scale); hi[3]=(__bf16)(a2.w*scale);
            hi[4]=(__bf16)(a3.x*scale); hi[5]=(__bf16)(a3.y*scale);
            hi[6]=(__bf16)(a3.z*scale); hi[7]=(__bf16)(a3.w*scale);
            *(v8bf*)&A_ls[m][cseg]     = lo;
            *(v8bf*)&A_ls[m][cseg + 8] = hi;
        }
        __syncthreads();   // B2

        // ---- GEMM1 per 16-row subtile: batch-load 8 fragments, then chain
        //      8 WMMAs (D->C accumulate chain, counting dscnt waits) ----
        #pragma unroll
        for (int mt = 0; mt < SUBT; ++mt) {
            const int rb = mt * 16;
            SCHED_FENCE();
            v16bf af[8];
            #pragma unroll
            for (int k = 0; k < 8; ++k) {
                const __bf16* pa = &A_ls[rb + lmod][k * 32 + lhi * 8];
                v8bf alo = *(const v8bf*)pa;
                v8bf ahi = *(const v8bf*)(pa + 16);
                af[k] = __builtin_shufflevector(alo, ahi,
                    0,1,2,3,4,5,6,7,8,9,10,11,12,13,14,15);
            }
            SCHED_FENCE();   // keep all 16 ds_loads in flight before first WMMA
            v8f acc;
            #pragma unroll
            for (int i = 0; i < 8; ++i) acc[i] = bias1;
            #pragma unroll
            for (int k = 0; k < 8; ++k)
                acc = __builtin_amdgcn_wmma_f32_16x16x32_bf16(
                    false, af[k], false, w1f[k], (short)0, acc, false, false);
            // ReLU (single v_max_num_f32) + stash hidden tile (bf16).
            // C/D layout: VGPR i, lane -> M = i + 8*lhi.
            #pragma unroll
            for (int i = 0; i < 8; ++i) {
                float h = fmaxf(acc[i], 0.0f);
                h_ls[rb + i + 8 * lhi][N] = (__bf16)h;
            }
            SCHED_FENCE();   // contain register pressure to one subtile
        }
        __syncthreads();   // B3

        // ---- GEMM2 + scatter per subtile ----
        #pragma unroll
        for (int mt = 0; mt < SUBT; ++mt) {
            const int rb = mt * 16;
            SCHED_FENCE();
            v16bf hf[4];
            #pragma unroll
            for (int k = 0; k < 4; ++k) {
                const __bf16* pa = &h_ls[rb + lmod][k * 32 + lhi * 8];
                v8bf alo = *(const v8bf*)pa;
                v8bf ahi = *(const v8bf*)(pa + 16);
                hf[k] = __builtin_shufflevector(alo, ahi,
                    0,1,2,3,4,5,6,7,8,9,10,11,12,13,14,15);
            }
            SCHED_FENCE();
            v8f acc2;
            #pragma unroll
            for (int i = 0; i < 8; ++i) acc2[i] = bias2;
            #pragma unroll
            for (int k = 0; k < 4; ++k)
                acc2 = __builtin_amdgcn_wmma_f32_16x16x32_bf16(
                    false, hf[k], false, w2f[k], (short)0, acc2, false, false);
            #pragma unroll
            for (int i = 0; i < 8; ++i) {
                int M = rb + i + 8 * lhi;
                if (ebase + M < E) {
                    float* dst = &out[(size_t)s_col[parity][M] * OUT_CH + N];
                    __hip_atomic_fetch_add(dst, acc2[i], __ATOMIC_RELAXED,
                                           __HIP_MEMORY_SCOPE_AGENT);
                }
            }
            SCHED_FENCE();
        }
        // no trailing barrier: s_col is double-buffered; B1 of the next
        // iteration orders all A_ls/h_ls rewrites after this body; s_row/s_ea are last read before B2.
    }
}

extern "C" void kernel_launch(void* const* d_in, const int* in_sizes, int n_in,
                              void* d_out, int out_size, void* d_ws, size_t ws_size,
                              hipStream_t stream) {
    const float* x   = (const float*)d_in[0];
    const int*   ei  = (const int*)  d_in[1];   // [2, E]
    const float* ea  = (const float*)d_in[2];   // [E]
    const float* W1  = (const float*)d_in[3];   // [256,128]
    const float* b1  = (const float*)d_in[4];
    const float* W2  = (const float*)d_in[5];   // [128,128]
    const float* b2  = (const float*)d_in[6];
    float* out = (float*)d_out;

    const int E       = in_sizes[2];
    const int n_iters = (E + TILE_E - 1) / TILE_E;

    zero_f32_kernel<<<(out_size + 255) / 256, 256, 0, stream>>>(out, out_size);

    int grid = n_iters < 960 ? n_iters : 960;
    edge_mlp_wmma_kernel<<<grid, BLOCK_T, 0, stream>>>(
        x, ei, ea, W1, b1, W2, b2, out, E, n_iters);
}
